// HungarianMatcher_14602888806441
// MI455X (gfx1250) — compile-verified, matrix operand order checked by reading, not run on hardware
//
#include <hip/hip_runtime.h>
#include <hip/hip_bf16.h>

// ---------------- constants (match reference) ----------------
constexpr float ALPHA_  = 0.25f;
constexpr float WCLASS  = 2.0f;
constexpr float WBBOX   = 5.0f;
constexpr float WGIOU   = 2.0f;
constexpr float EPSLOG  = 1e-8f;
constexpr float EPSDIV  = 1e-6f;

constexpr int Bn   = 16;
constexpr int Qn   = 900;
constexpr int Cn   = 91;
constexpr int Tn   = 4800;
constexpr int BQ   = Bn * Qn;     // 14400
constexpr int KPAD = 96;          // pad 91 -> 96 = 3 * 32 (bf16 WMMA K)

typedef __attribute__((ext_vector_type(16))) __bf16       v16bf;
typedef __attribute__((ext_vector_type(8)))  __bf16       bf16x8;
typedef __attribute__((ext_vector_type(8)))  float        v8f;
typedef __attribute__((ext_vector_type(4)))  float        f4;
typedef __attribute__((ext_vector_type(4)))  unsigned int u32x4;
typedef __attribute__((ext_vector_type(8)))  int          i32x8;
typedef __attribute__((ext_vector_type(4)))  int          i32x4;

__device__ __forceinline__ float rcp_fast(float x) { return __builtin_amdgcn_rcpf(x); }

#if defined(__gfx1250__) && __has_builtin(__builtin_amdgcn_tensor_load_to_lds)
#define HAVE_TDM 1
#else
#define HAVE_TDM 0
#endif

#if HAVE_TDM
// 1-D TDM DMA: copy n8 * 8 bytes from global -> LDS via the Tensor Data Mover.
// D# layout per CDNA5 ISA 8.3/8.4: group0 = {flags, lds_addr, global_addr, type=2},
// group1 = {data_size=8B, tensor_dim0=n8, tensor_dim1=1, tile_dim0=n8, tile_dim1=1,
//           tensor_dim0_stride=n8}. Groups 2/3 (and trailing group) zero (<=2D tensor).
__device__ __forceinline__ void tdm_load_1d(void* lds, const void* gsrc, unsigned n8) {
    unsigned           laddr = (unsigned)(unsigned long long)(uintptr_t)lds;
    unsigned long long ga    = (unsigned long long)(uintptr_t)gsrc;

    u32x4 g0;
    g0[0] = 1u;                                             // count=1, user mode
    g0[1] = laddr;                                          // lds_addr [63:32]
    g0[2] = (unsigned)(ga & 0xFFFFFFFFull);                 // global_addr [95:64]
    g0[3] = (unsigned)((ga >> 32) & 0x01FFFFFFull)          // global_addr [120:96]
          | (2u << 30);                                     // type=2 ("image")

    i32x8 g1 = (i32x8){0, 0, 0, 0, 0, 0, 0, 0};
    g1[0] = (3 << 16);                                      // data_size = 8 bytes
    g1[1] = (int)((n8 & 0xFFFFu) << 16);                    // tensor_dim0 [63:48]
    g1[2] = (int)(((n8 >> 16) & 0xFFFFu)                    // tensor_dim0 [79:64]
          | (1u << 16));                                    // tensor_dim1 = 1 [111:80]
    g1[3] = (int)((n8 & 0xFFFFu) << 16);                    // tile_dim0 [127:112]
    g1[4] = 1;                                              // tile_dim1 = 1 [143:128]
    g1[5] = (int)n8;                                        // tensor_dim0_stride [207:160]

    i32x4 z4 = (i32x4){0, 0, 0, 0};
    i32x8 z8 = (i32x8){0, 0, 0, 0, 0, 0, 0, 0};
    __builtin_amdgcn_tensor_load_to_lds(g0, g1, z4, z4, z8, 0);  // 6-arg form
}
#endif

// ---------------- prep: focal class-cost delta, split into bf16 hi+lo ----------------
__global__ __launch_bounds__(256) void prep_delta(const float* __restrict__ logits,
                                                  __bf16* __restrict__ hi,
                                                  __bf16* __restrict__ lo) {
    int idx = blockIdx.x * 256 + threadIdx.x;
    if (idx >= BQ * KPAD) return;
    int col = idx % KPAD;
    int row = idx / KPAD;
    float d = 0.0f;
    if (col < Cn) {
        float x  = logits[row * Cn + col];
        float pl = 1.0f / (1.0f + __expf(-x));          // sigmoid
        float om = 1.0f - pl;
        float neg = (1.0f - ALPHA_) * pl * pl * (-__logf(om + EPSLOG));
        float pos = ALPHA_ * om * om * (-__logf(pl + EPSLOG));
        d = WCLASS * (pos - neg);
    }
    __bf16 h = (__bf16)d;
    __bf16 l = (__bf16)(d - (float)h);
    hi[idx] = h;
    lo[idx] = l;
}

// ---------------- prep: box records {cx,cy,w,h, x0,y0,x1,y1, area, pad x3} ----------------
__global__ __launch_bounds__(256) void prep_boxes(const float* __restrict__ boxes,
                                                  float* __restrict__ rec, int n) {
    int i = blockIdx.x * 256 + threadIdx.x;
    if (i >= n) return;
    float cx = boxes[i * 4 + 0], cy = boxes[i * 4 + 1];
    float w  = boxes[i * 4 + 2], h  = boxes[i * 4 + 3];
    float x0 = cx - 0.5f * w, y0 = cy - 0.5f * h;
    float x1 = cx + 0.5f * w, y1 = cy + 0.5f * h;
    float area = (x1 - x0) * (y1 - y0);
    float* p = rec + i * 12;
    p[0] = cx; p[1] = cy; p[2] = w;  p[3] = h;
    p[4] = x0; p[5] = y0; p[6] = x1; p[7] = y1;
    p[8] = area; p[9] = 0.0f; p[10] = 0.0f; p[11] = 0.0f;
}

__global__ __launch_bounds__(256) void prep_cls(const int* __restrict__ ids,
                                                int* __restrict__ cls, int n) {
    int i = blockIdx.x * 256 + threadIdx.x;
    if (i >= n) return;
    cls[i] = ids[i] - 1;                                 // 0..90
}

// ---------------- main: 64x64 output tile per block ----------------
__global__ __launch_bounds__(256) void match_cost(const __bf16* __restrict__ dHi,
                                                  const __bf16* __restrict__ dLo,
                                                  const float* __restrict__ rowrec,
                                                  const float* __restrict__ tgtrec,
                                                  const int* __restrict__ cls,
                                                  float* __restrict__ out) {
    __shared__ __align__(16) __bf16 sHi[64 * KPAD];      // 12 KB
    __shared__ __align__(16) __bf16 sLo[64 * KPAD];      // 12 KB
    __shared__ __align__(16) float  sRow[64 * 12];       // 3 KB
    __shared__ __align__(16) float  sTgt[64 * 12];       // 3 KB
    __shared__ __align__(16) int    sCls[64];

    const int tid = threadIdx.x;
    const int bx  = blockIdx.x;   // target tile (75)
    const int by  = blockIdx.y;   // row tile (225)
    const int wv  = tid >> 5;

    // ---- stage tile into LDS ----
#if HAVE_TDM
    {
        if (wv == 0)      tdm_load_1d(sHi,  dHi + (size_t)by * 64 * KPAD, 64 * KPAD / 4);
        else if (wv == 1) tdm_load_1d(sLo,  dLo + (size_t)by * 64 * KPAD, 64 * KPAD / 4);
        else if (wv == 2) tdm_load_1d(sRow, rowrec + (size_t)by * 64 * 12, 64 * 12 / 2);
        else if (wv == 3) tdm_load_1d(sTgt, tgtrec + (size_t)bx * 64 * 12, 64 * 12 / 2);
        else if (wv == 4) tdm_load_1d(sCls, cls + bx * 64, 64 / 2);
        __builtin_amdgcn_s_wait_tensorcnt(0);
    }
#else
    {
        const u32x4* gh = (const u32x4*)(dHi + (size_t)by * 64 * KPAD);
        const u32x4* gl = (const u32x4*)(dLo + (size_t)by * 64 * KPAD);
        u32x4* sh = (u32x4*)sHi;
        u32x4* sl = (u32x4*)sLo;
#pragma unroll
        for (int k = 0; k < 3; ++k) {
            sh[tid + 256 * k] = gh[tid + 256 * k];
            sl[tid + 256 * k] = gl[tid + 256 * k];
        }
        const u32x4* gr = (const u32x4*)(rowrec + (size_t)by * 64 * 12);
        const u32x4* gt = (const u32x4*)(tgtrec + (size_t)bx * 64 * 12);
        if (tid < 192) {
            ((u32x4*)sRow)[tid] = gr[tid];
            ((u32x4*)sTgt)[tid] = gt[tid];
        }
        if (tid < 64) sCls[tid] = cls[bx * 64 + tid];
    }
#endif
    __syncthreads();

    const int  lane = tid & 31;
    const int  ln   = lane & 15;
    const bool up   = (lane >= 16);
    const int  koff = up ? 8 : 0;        // K half-select (A/B 16-bit WMMA layout)
    const int  roff = up ? 8 : 0;        // C/D: lanes 16-31 hold rows M+8

    // wave handles tiles (rbase, c0) and (rbase+32, c0): same columns, rows +32
    const int c0    = (wv & 3) * 16;
    const int rbase = (wv >> 2) * 16;
    const int n     = c0 + ln;

    // ---- target record (shared by both tiles) ----
    const f4* tp = (const f4*)&sTgt[n * 12];
    const f4 t0 = tp[0], t1 = tp[1], t2 = tp[2];
    const float tcx = t0[0], tcy = t0[1], tw = t0[2], th = t0[3];
    const float tx0 = t1[0], ty0 = t1[1], tx1 = t1[2], ty1 = t1[3];
    const float tA  = t2[0];

    // ---- one-hot B (bf16), built once, reused by all 4 WMMA chains x 3 chunks ----
    const int cn = sCls[n];
    v16bf Bv[3];
#pragma unroll
    for (int ch = 0; ch < 3; ++ch) {
        const int kc = ch * 32;
#pragma unroll
        for (int j = 0; j < 4; ++j) {
            const int ka = kc + koff + 2 * j;
            const int kb = kc + 16 + koff + 2 * j;
            Bv[ch][2 * j]         = (cn == ka)     ? (__bf16)1.0f : (__bf16)0.0f;
            Bv[ch][2 * j + 1]     = (cn == ka + 1) ? (__bf16)1.0f : (__bf16)0.0f;
            Bv[ch][8 + 2 * j]     = (cn == kb)     ? (__bf16)1.0f : (__bf16)0.0f;
            Bv[ch][8 + 2 * j + 1] = (cn == kb + 1) ? (__bf16)1.0f : (__bf16)0.0f;
        }
    }

    // ---- VALU: bbox + giou terms for both row groups ----
    v8f acc0 = {}, acc1 = {};
#pragma unroll
    for (int i = 0; i < 8; ++i) {
#pragma unroll
        for (int half = 0; half < 2; ++half) {
            const int m = rbase + i + roff + 32 * half;
            const f4* rp = (const f4*)&sRow[m * 12];     // broadcast read
            const f4 a0 = rp[0], a1 = rp[1], a2 = rp[2];
            const float rcx = a0[0], rcy = a0[1], rw = a0[2], rh = a0[3];
            const float rx0 = a1[0], ry0 = a1[1], rx1 = a1[2], ry1 = a1[3];
            const float rA  = a2[0];

            float iw    = fmaxf(fminf(rx1, tx1) - fmaxf(rx0, tx0), 0.0f);
            float ih    = fmaxf(fminf(ry1, ty1) - fmaxf(ry0, ty0), 0.0f);
            float inter = iw * ih;
            float uni   = rA + tA - inter;
            float iou   = inter * rcp_fast(fmaxf(uni, EPSDIV));
            float ew    = fmaxf(fmaxf(rx1, tx1) - fminf(rx0, tx0), 0.0f);
            float eh    = fmaxf(fmaxf(ry1, ty1) - fminf(ry0, ty0), 0.0f);
            float encl  = ew * eh;
            float giou  = iou - (encl - uni) * rcp_fast(fmaxf(encl, EPSDIV));
            float l1    = fabsf(rcx - tcx) + fabsf(rcy - tcy)
                        + fabsf(rw - tw)   + fabsf(rh - th);
            float v = WBBOX * l1 - WGIOU * giou;
            if (half == 0) acc0[i] = v; else acc1[i] = v;
        }
    }

    // ---- WMMA gather: two independent accumulator chains, interleaved ----
    const __bf16* hr0 = &sHi[(rbase + ln) * KPAD];
    const __bf16* lr0 = &sLo[(rbase + ln) * KPAD];
    const __bf16* hr1 = hr0 + 32 * KPAD;
    const __bf16* lr1 = lr0 + 32 * KPAD;
#pragma unroll
    for (int ch = 0; ch < 3; ++ch) {
        const int kc = ch * 32;
        bf16x8 h0a = *(const bf16x8*)(hr0 + kc + koff);
        bf16x8 h0b = *(const bf16x8*)(hr0 + kc + 16 + koff);
        bf16x8 l0a = *(const bf16x8*)(lr0 + kc + koff);
        bf16x8 l0b = *(const bf16x8*)(lr0 + kc + 16 + koff);
        bf16x8 h1a = *(const bf16x8*)(hr1 + kc + koff);
        bf16x8 h1b = *(const bf16x8*)(hr1 + kc + 16 + koff);
        bf16x8 l1a = *(const bf16x8*)(lr1 + kc + koff);
        bf16x8 l1b = *(const bf16x8*)(lr1 + kc + 16 + koff);
        v16bf aH0, aL0, aH1, aL1;
#pragma unroll
        for (int e = 0; e < 8; ++e) {
            aH0[e] = h0a[e]; aH0[8 + e] = h0b[e];
            aL0[e] = l0a[e]; aL0[8 + e] = l0b[e];
            aH1[e] = h1a[e]; aH1[8 + e] = h1b[e];
            aL1[e] = l1a[e]; aL1[8 + e] = l1b[e];
        }
        // interleaved chains: dependent WMMAs separated by an independent one
        acc0 = __builtin_amdgcn_wmma_f32_16x16x32_bf16(false, aL0, false, Bv[ch],
                                                       (short)0, acc0, false, false);
        acc1 = __builtin_amdgcn_wmma_f32_16x16x32_bf16(false, aL1, false, Bv[ch],
                                                       (short)0, acc1, false, false);
        acc0 = __builtin_amdgcn_wmma_f32_16x16x32_bf16(false, aH0, false, Bv[ch],
                                                       (short)0, acc0, false, false);
        acc1 = __builtin_amdgcn_wmma_f32_16x16x32_bf16(false, aH1, false, Bv[ch],
                                                       (short)0, acc1, false, false);
    }

    // ---- store both 16x16 tiles ----
    float* ob0 = out + (size_t)(by * 64 + rbase + roff) * Tn + (size_t)(bx * 64 + n);
    float* ob1 = ob0 + (size_t)32 * Tn;
#pragma unroll
    for (int i = 0; i < 8; ++i) {
        ob0[(size_t)i * Tn] = acc0[i];
        ob1[(size_t)i * Tn] = acc1[i];
    }
}

// ---------------- launch ----------------
extern "C" void kernel_launch(void* const* d_in, const int* in_sizes, int n_in,
                              void* d_out, int out_size, void* d_ws, size_t ws_size,
                              hipStream_t stream) {
    const float* logits = (const float*)d_in[0];   // [16,900,91]
    const float* pboxes = (const float*)d_in[1];   // [16,900,4]
    const int*   tids   = (const int*)d_in[2];     // [4800]
    const float* tboxes = (const float*)d_in[3];   // [4800,4]
    float* out = (float*)d_out;

    char* ws = (char*)d_ws;
    __bf16* dHi   = (__bf16*)(ws);                 // 2,764,800 B
    __bf16* dLo   = (__bf16*)(ws + 2764800);       // 2,764,800 B
    float*  rowrc = (float*)(ws + 5529600);        // 691,200 B
    float*  tgtrc = (float*)(ws + 6220800);        // 230,400 B
    int*    clsb  = (int*)(ws + 6451200);          // 19,200 B

    prep_delta<<<(BQ * KPAD + 255) / 256, 256, 0, stream>>>(logits, dHi, dLo);
    prep_boxes<<<(BQ + 255) / 256, 256, 0, stream>>>(pboxes, rowrc, BQ);
    prep_boxes<<<(Tn + 255) / 256, 256, 0, stream>>>(tboxes, tgtrc, Tn);
    prep_cls<<<(Tn + 255) / 256, 256, 0, stream>>>(tids, clsb, Tn);

    dim3 grid(Tn / 64, BQ / 64);                   // 75 x 225 blocks
    match_cost<<<grid, 256, 0, stream>>>(dHi, dLo, rowrc, tgtrc, clsb, out);
}
